// UserEncoder_66614942761558
// MI455X (gfx1250) — matrix operand order repaired and checked by reading, not compiled, not run on hardware
//
#include <hip/hip_runtime.h>
#include <hip/hip_bf16.h>
#include <cfloat>

typedef __attribute__((ext_vector_type(16))) _Float16 v16h;
typedef __attribute__((ext_vector_type(8)))  float    v8f;

#define BB   32
#define HH   100
#define CC   50
#define DD   400
#define HIDN 400

// ---------------- Kernel 0: v[d] = sum_k W1h[d,k] * w2[k] ----------------
__global__ void __launch_bounds__(256)
ue_reduce_w1h_kernel(const float* __restrict__ W1,
                     const float* __restrict__ w2,
                     float* __restrict__ v) {
    int d = blockIdx.x * blockDim.x + threadIdx.x;
    if (d >= DD) return;
    const float* row = W1 + (size_t)(DD + d) * HIDN;   // W1h = W1[D:2D, :]
    float acc = 0.f;
    #pragma unroll 4
    for (int k = 0; k < HIDN; ++k) acc = fmaf(row[k], w2[k], acc);
    v[d] = acc;
}

// Load 8 consecutive f32 from a row (base is a multiple of 8) as f16, zero if invalid.
__device__ inline void load8_h(const float* __restrict__ rowptr, int base, bool valid,
                               _Float16* __restrict__ dst) {
    if (valid) {
        float4 x = *(const float4*)(rowptr + base);
        float4 y = *(const float4*)(rowptr + base + 4);
        dst[0] = (_Float16)x.x; dst[1] = (_Float16)x.y;
        dst[2] = (_Float16)x.z; dst[3] = (_Float16)x.w;
        dst[4] = (_Float16)y.x; dst[5] = (_Float16)y.y;
        dst[6] = (_Float16)y.z; dst[7] = (_Float16)y.w;
    } else {
        #pragma unroll
        for (int i = 0; i < 8; ++i) dst[i] = (_Float16)0.f;
    }
}

// ---------------- Kernel 1: per-batch scores (WMMA) + softmax + u ----------------
__global__ void __launch_bounds__(256)
ue_attend_kernel(const float* __restrict__ hist,
                 const int*   __restrict__ mask_hist,
                 const float* __restrict__ v,
                 float*       __restrict__ u_ws) {
    __shared__ float s_v[416];    // v padded to 13*32 with zeros
    __shared__ float s_s[128];    // scores, then softmax weights

    const int b    = blockIdx.x;
    const int tid  = threadIdx.x;
    const int lane = tid & 31;
    const int wave = tid >> 5;
    const float* hist_b = hist + (size_t)b * HH * DD;

    for (int i = tid; i < 416; i += 256) s_v[i] = (i < DD) ? v[i] : 0.f;
    __syncthreads();

    // ---- Phase A: scores[h] = hist[b,h,:] . v  via v_wmma_f32_16x16x32_f16 ----
    if (wave < 7) {                       // 7 tiles of 16 rows cover H=100
        const int  m0     = wave * 16;
        const int  m      = m0 + (lane & 15);
        const bool rowv   = (m < HH);
        const bool hiHalf = (lane >= 16);
        const float* rp   = hist_b + (size_t)m * DD;

        v8f c = {};
        for (int k0 = 0; k0 < DD; k0 += 32) {
            // A fragment: lanes 0-15 hold K {k0..k0+7, k0+16..k0+23},
            //             lanes 16-31 hold K {k0+8..k0+15, k0+24..k0+31}
            _Float16 tmp[16];
            int g0 = k0 + (hiHalf ? 8  : 0);
            int g1 = k0 + (hiHalf ? 24 : 16);
            load8_h(rp, g0, rowv && (g0 + 8 <= DD), tmp);
            load8_h(rp, g1, rowv && (g1 + 8 <= DD), tmp + 8);
            v16h a;
            #pragma unroll
            for (int i = 0; i < 16; ++i) a[i] = tmp[i];

            // B fragment: column N=0 holds v; lane 0 -> K {k0..k0+7, k0+16..k0+23}, lane 16 -> K {k0+8..k0+15, k0+24..k0+31}
            v16h bf = {};
            if (lane == 0 || lane == 16) {
                int kb = k0 + (lane == 0 ? 0 : 8);
                #pragma unroll
                for (int i = 0; i < 16; ++i) bf[i] = (_Float16)s_v[kb + i + (i >= 8 ? 8 : 0)];
            }
            c = __builtin_amdgcn_wmma_f32_16x16x32_f16(
                    false, a, false, bf, (short)0, c, false, false);
        }
        // C/D layout: lane 0 holds (M=j, N=0) in c[j]; lane 16 holds (M=8+j, N=0)
        if (lane == 0) {
            #pragma unroll
            for (int j = 0; j < 8; ++j) { int h = m0 + j;     if (h < HH) s_s[h] = c[j]; }
        } else if (lane == 16) {
            #pragma unroll
            for (int j = 0; j < 8; ++j) { int h = m0 + 8 + j; if (h < HH) s_s[h] = c[j]; }
        }
    }
    __syncthreads();

    // ---- Phase B: masked softmax over h (wave 0) ----
    if (wave == 0) {
        float vals[4];
        float mx = -FLT_MAX;                         // matches jnp.finfo(f32).min
        #pragma unroll
        for (int i = 0; i < 4; ++i) {
            int h = lane + 32 * i;
            float s = -FLT_MAX;
            if (h < HH) s = (mask_hist[b * HH + h] != 0) ? s_s[h] : -FLT_MAX;
            vals[i] = s;
            mx = fmaxf(mx, s);
        }
        #pragma unroll
        for (int off = 16; off > 0; off >>= 1) mx = fmaxf(mx, __shfl_xor(mx, off, 32));
        float e[4], sum = 0.f;
        #pragma unroll
        for (int i = 0; i < 4; ++i) {
            int h = lane + 32 * i;
            e[i] = (h < HH) ? __expf(vals[i] - mx) : 0.f;   // all-masked row -> uniform 1/H
            sum += e[i];
        }
        #pragma unroll
        for (int off = 16; off > 0; off >>= 1) sum += __shfl_xor(sum, off, 32);
        float inv = 1.f / sum;
        #pragma unroll
        for (int i = 0; i < 4; ++i) {
            int h = lane + 32 * i;
            if (h < HH) s_s[h] = e[i] * inv;
        }
    }
    __syncthreads();

    // ---- Phase C: u[d] = sum_h w[h] * hist[b,h,d]  (hist[b] is L2-resident now) ----
    for (int d = tid; d < DD; d += 256) {
        float acc = 0.f;
        for (int h = 0; h < HH; ++h) {
            if (h + 4 < HH) __builtin_prefetch(hist_b + (size_t)(h + 4) * DD + d, 0, 0);
            acc = fmaf(s_s[h], hist_b[(size_t)h * DD + d], acc);
        }
        u_ws[b * DD + d] = acc;
    }
}

// ---------------- Kernel 2: out[b,c,:] = mask_cand[b,c] ? u[b,:] : 0 ----------------
__global__ void __launch_bounds__(256)
ue_broadcast_kernel(const float* __restrict__ u_ws,
                    const int*   __restrict__ mask_cand,
                    float*       __restrict__ out) {
    int b = blockIdx.y, c = blockIdx.x;
    bool mc = (mask_cand[b * CC + c] != 0);
    const float* u = u_ws + b * DD;
    float* o = out + ((size_t)b * CC + c) * DD;
    for (int d = threadIdx.x; d < DD; d += 256)
        o[d] = mc ? u[d] : 0.f;
}

extern "C" void kernel_launch(void* const* d_in, const int* in_sizes, int n_in,
                              void* d_out, int out_size, void* d_ws, size_t ws_size,
                              hipStream_t stream) {
    (void)in_sizes; (void)n_in; (void)out_size; (void)ws_size;
    const float* hist      = (const float*)d_in[0];
    // d_in[1] cand_news_vector: unused — its score contribution is constant over h
    // and cancels exactly in the softmax (as do b1 = d_in[5] and b2 = d_in[7]).
    const int*   mask_hist = (const int*)d_in[2];
    const int*   mask_cand = (const int*)d_in[3];
    const float* W1        = (const float*)d_in[4];
    const float* w2        = (const float*)d_in[6];
    float* out = (float*)d_out;

    float* wsf  = (float*)d_ws;
    float* v_ws = wsf;           // 400 floats
    float* u_ws = wsf + 512;     // 32*400 floats

    ue_reduce_w1h_kernel<<<dim3(2),       dim3(256), 0, stream>>>(W1, w2, v_ws);
    ue_attend_kernel   <<<dim3(BB),      dim3(256), 0, stream>>>(hist, mask_hist, v_ws, u_ws);
    ue_broadcast_kernel<<<dim3(CC, BB),  dim3(256), 0, stream>>>(u_ws, mask_cand, out);
}
